// Block_86912958201919
// MI455X (gfx1250) — matrix operand order repaired and checked by reading, not compiled, not run on hardware
//
#include <hip/hip_runtime.h>
#include <hip/hip_bf16.h>

typedef __attribute__((ext_vector_type(16))) _Float16 v16h;
typedef __attribute__((ext_vector_type(8)))  float    v8f;
typedef _Float16 half_t;

#define B_   2
#define T_   2048
#define C_   1024
#define H_   16
#define HS_  64
#define FF_  4096
#define BT_  (B_ * T_)
#define QKVLD 3072          // packed q|k|v column width

static __device__ inline v8f zero8() {
  v8f z = {0.f,0.f,0.f,0.f,0.f,0.f,0.f,0.f};
  return z;
}

static __device__ inline v8f wmma_f16(v16h a, v16h b, v8f c) {
  // D = A(16x32 f16) * B(32x16 f16) + C(16x16 f32)
  return __builtin_amdgcn_wmma_f32_16x16x32_f16(
      /*neg_a=*/false, a, /*neg_b=*/false, b,
      /*c_mod=*/(short)0, c, /*reuse_a=*/false, /*reuse_b=*/false);
}

// Load a v16h fragment from two contiguous 16-byte chunks.
static __device__ inline v16h ld_frag2(const half_t* p0, const half_t* p1) {
  union { v16h v; uint4 u[2]; } r;
  r.u[0] = *(const uint4*)p0;
  r.u[1] = *(const uint4*)p1;
  return r.v;
}

// Async copy 16B global -> LDS (CDNA5 GLOBAL_LOAD_ASYNC_TO_LDS_B128, ASYNCcnt).
// LDS byte offset = low 32 bits of the generic shared pointer (aperture rule).
static __device__ inline void async_g2l_b128(const half_t* gptr, half_t* lptr) {
  unsigned int lds_off = (unsigned int)(unsigned long long)lptr;
  asm volatile("global_load_async_to_lds_b128 %0, %1, off"
               :: "v"(lds_off), "v"(gptr) : "memory");
}
static __device__ inline void wait_asynccnt0() {
  asm volatile("s_wait_asynccnt 0x0" ::: "memory");
}

// ---------------------------------------------------------------- converts
__global__ void cvt_f32_f16(const float* __restrict__ src,
                            half_t* __restrict__ dst, int n) {
  int i = blockIdx.x * blockDim.x + threadIdx.x;
  if (i < n) dst[i] = (half_t)src[i];
}

// wq/wk/wv (H,C,HS) fp32 -> packed (C, 3*C) f16, col = kind*C + h*HS + d
__global__ void pack_wqkv(const float* __restrict__ wq,
                          const float* __restrict__ wk,
                          const float* __restrict__ wv,
                          half_t* __restrict__ dst) {
  int i = blockIdx.x * blockDim.x + threadIdx.x;
  if (i >= 3 * C_ * C_) return;
  int kind = i / (C_ * C_);
  int rem  = i % (C_ * C_);
  int c    = rem / C_;
  int col  = rem % C_;          // h*64 + d
  int h = col >> 6, d = col & 63;
  const float* w = (kind == 0) ? wq : (kind == 1) ? wk : wv;
  dst[(size_t)c * QKVLD + kind * C_ + col] = (half_t)w[((size_t)h * C_ + c) * HS_ + d];
}

// ---------------------------------------------------------------- GEMM
// C(MxN,f32 acc) = A(MxK,f16,row-major) * B(KxN,f16,row-major) [+bias][relu]
// Block tile 128x128, K-step 64; 256 threads = 8 waves (4x2), wave tile 32x64.
// 16 WMMAs per LDS stage; A staged via async-to-LDS, B staged transposed.
#define BM 128
#define BN 128
#define BK 64

__global__ __launch_bounds__(256)
void gemm_wmma(const half_t* __restrict__ A, int lda,
               const half_t* __restrict__ Bm, int ldb,
               const float* __restrict__ bias,
               float* __restrict__ C32, half_t* __restrict__ C16, int ldc,
               int K, int do_relu) {
  __shared__ half_t As[BM * BK];   // row-major rows, stride BK      (16 KB)
  __shared__ half_t Bs[BN * BK];   // TRANSPOSED: Bs[n*BK + k]       (16 KB)

  const int tid  = threadIdx.x;
  const int wid  = tid >> 5;
  const int lane = tid & 31;
  const int ln   = lane & 15;      // fragment lane index
  const int lg   = lane >> 4;      // lane group (0/1)
  const int wm   = wid >> 1;       // 0..3 (M direction)
  const int wn   = wid & 1;        // 0..1 (N direction)
  const int rowBlock = blockIdx.y * BM;
  const int colBlock = blockIdx.x * BN;

  v8f acc[2][4];
#pragma unroll
  for (int i = 0; i < 2; ++i)
#pragma unroll
    for (int j = 0; j < 4; ++j) acc[i][j] = zero8();

  for (int k0 = 0; k0 < K; k0 += BK) {
    // ---- stage A tile (128x64 halves = 16KB) via async global->LDS b128.
    // 1024 chunks of 8 halves; 4 per thread; coalesced 128B per 8 lanes.
#pragma unroll
    for (int it = 0; it < 4; ++it) {
      int chunk = tid + it * 256;
      int r  = chunk >> 3;               // 0..127
      int cc = chunk & 7;                // 0..7
      const half_t* g = A + (size_t)(rowBlock + r) * lda + k0 + cc * 8;
      async_g2l_b128(g, &As[r * BK + cc * 8]);
      if (k0 + BK < K) __builtin_prefetch(g + BK, 0, 1);
    }
    // ---- stage B tile (64x128 halves) transposed into Bs[n][k]
#pragma unroll
    for (int it = 0; it < 4; ++it) {
      int chunk = tid + it * 256;        // 1024 chunks of 8 halves
      int kr = chunk >> 4;               // 0..63
      int nc = chunk & 15;               // 0..15
      union { uint4 u; half_t h[8]; } v;
      const half_t* g = Bm + (size_t)(k0 + kr) * ldb + colBlock + nc * 8;
      v.u = *(const uint4*)g;
      if (k0 + BK < K) __builtin_prefetch(g + (size_t)BK * ldb, 0, 1);
#pragma unroll
      for (int j = 0; j < 8; ++j) Bs[(nc * 8 + j) * BK + kr] = v.h[j];
    }
    wait_asynccnt0();     // this wave's async A-loads have landed in LDS
    __syncthreads();      // + dscnt fence publishes B tile workgroup-wide

    // ---- 2 k-steps of 32, fragments per ISA 7.12.2 layouts
#pragma unroll
    for (int ks = 0; ks < 2; ++ks) {
      v16h af[2], bf[4];
#pragma unroll
      for (int mt = 0; mt < 2; ++mt) {
        const half_t* p = &As[(wm * 32 + mt * 16 + ln) * BK + ks * 32 + 8 * lg];
        af[mt] = ld_frag2(p, p + 16);    // K = 8g..8g+7 , 16+8g..16+8g+7
      }
#pragma unroll
      for (int nt = 0; nt < 4; ++nt) {
        const half_t* p = &Bs[(wn * 64 + nt * 16 + ln) * BK + ks * 32 + 8 * lg];
        bf[nt] = ld_frag2(p, p + 16);    // K = 8g..8g+7 , 16+8g..16+8g+7
      }
#pragma unroll
      for (int mt = 0; mt < 2; ++mt)
#pragma unroll
        for (int nt = 0; nt < 4; ++nt)
          acc[mt][nt] = wmma_f16(af[mt], bf[nt], acc[mt][nt]);
    }
    __syncthreads();
  }

  // ---- epilogue: element r of acc <-> row (r + 8*lg), lane ln <-> col
#pragma unroll
  for (int mt = 0; mt < 2; ++mt) {
#pragma unroll
    for (int nt = 0; nt < 4; ++nt) {
      int col = colBlock + wn * 64 + nt * 16 + ln;
      float bv = bias ? bias[col] : 0.f;
#pragma unroll
      for (int r = 0; r < 8; ++r) {
        int row = rowBlock + wm * 32 + mt * 16 + r + 8 * lg;
        float v = acc[mt][nt][r] + bv;
        if (do_relu && v < 0.f) v = 0.f;
        if (C16) C16[(size_t)row * ldc + col] = (half_t)v;
        else     C32[(size_t)row * ldc + col] = v;
      }
    }
  }
}

// ---------------------------------------------------------------- attention
// One wave per (b,h,16-row Q tile). Flash softmax, WMMA for QK^T and P*V.
__global__ __launch_bounds__(32)
void attn_flash(const half_t* __restrict__ qkv, half_t* __restrict__ out) {
  __shared__ half_t Ps[16 * 32];   // probs tile, row-major (rows x 32 cols)
  __shared__ half_t Vt[64 * 32];   // transposed V tile: Vt[d*32 + ss]

  const int lane = threadIdx.x;
  const int ln = lane & 15, lg = lane >> 4;
  const int t0 = blockIdx.x * 16;
  const int bh = blockIdx.y;
  const int b = bh >> 4, h = bh & 15;

  const half_t* qbase = qkv + (size_t)(b * T_) * QKVLD + h * HS_;
  const half_t* kbase = qbase + C_;
  const half_t* vbase = qbase + 2 * C_;

  // Q A-fragments for k-steps 0 and 1 (HS=64 -> 2 x K=32)
  v16h aq[2];
#pragma unroll
  for (int ks = 0; ks < 2; ++ks) {
    const half_t* p = qbase + (size_t)(t0 + ln) * QKVLD + ks * 32 + 8 * lg;
    aq[ks] = ld_frag2(p, p + 16);
  }

  float m[8], l[8];
  v8f o[4];
#pragma unroll
  for (int r = 0; r < 8; ++r) { m[r] = -3.0e38f; l[r] = 0.f; }
#pragma unroll
  for (int dt = 0; dt < 4; ++dt) o[dt] = zero8();

  const float scale = 0.125f;      // 1/sqrt(64)
  const int send = t0 + 16;        // causal bound (exclusive)

  for (int s0 = 0; s0 < send; s0 += 32) {
    // ---- scores S = Q K^T for 2 column sub-tiles of 16
    v8f sc[2];
#pragma unroll
    for (int ct = 0; ct < 2; ++ct) {
      sc[ct] = zero8();
#pragma unroll
      for (int ks = 0; ks < 2; ++ks) {
        // B-frag of K^T: row-major K rows are already "transposed B" layout
        const half_t* p = kbase + (size_t)(s0 + ct * 16 + ln) * QKVLD + ks * 32 + 8 * lg;
        v16h bk = ld_frag2(p, p + 16);
        sc[ct] = wmma_f16(aq[ks], bk, sc[ct]);
      }
    }

    // ---- online softmax (row r+8*lg owned by 16 lanes of group lg)
#pragma unroll
    for (int r = 0; r < 8; ++r) {
      int row = t0 + r + 8 * lg;
      float v0 = sc[0][r] * scale;
      float v1 = sc[1][r] * scale;
      if (s0 + ln > row)      v0 = -1.0e30f;
      if (s0 + 16 + ln > row) v1 = -1.0e30f;
      float mx = fmaxf(v0, v1);
#pragma unroll
      for (int d = 8; d >= 1; d >>= 1) mx = fmaxf(mx, __shfl_xor(mx, d, 32));
      float mn = fmaxf(m[r], mx);
      float alpha = __expf(m[r] - mn);
      float p0 = __expf(v0 - mn);
      float p1 = __expf(v1 - mn);
      float rs = p0 + p1;
#pragma unroll
      for (int d = 8; d >= 1; d >>= 1) rs += __shfl_xor(rs, d, 32);
      l[r] = l[r] * alpha + rs;
      o[0][r] *= alpha; o[1][r] *= alpha; o[2][r] *= alpha; o[3][r] *= alpha;
      m[r] = mn;
      Ps[(r + 8 * lg) * 32 + ln]      = (half_t)p0;
      Ps[(r + 8 * lg) * 32 + 16 + ln] = (half_t)p1;
    }

    // ---- stage V tile transposed: each lane copies one V row (64 halves)
    {
      const half_t* vp = vbase + (size_t)(s0 + lane) * QKVLD;
#pragma unroll
      for (int c8 = 0; c8 < 8; ++c8) {
        union { uint4 u; half_t h[8]; } vv;
        vv.u = *(const uint4*)(vp + c8 * 8);
#pragma unroll
        for (int j = 0; j < 8; ++j) Vt[(c8 * 8 + j) * 32 + lane] = vv.h[j];
      }
    }

    // ---- O += P * V  (K=32, N=64 -> 4 WMMAs); same-wave DS ops are in-order
    {
      const half_t* pp = &Ps[ln * 32 + 8 * lg];
      v16h pf = ld_frag2(pp, pp + 16);
#pragma unroll
      for (int dt = 0; dt < 4; ++dt) {
        const half_t* vp = &Vt[(dt * 16 + ln) * 32 + 8 * lg];
        v16h bv = ld_frag2(vp, vp + 16);
        o[dt] = wmma_f16(pf, bv, o[dt]);
      }
    }
  }

  // ---- normalize and store f16 attn output at (b*T+row, h*64 + d)
#pragma unroll
  for (int r = 0; r < 8; ++r) {
    float inv = 1.f / l[r];
    int row = t0 + r + 8 * lg;
    half_t* op = out + (size_t)(b * T_ + row) * C_ + h * HS_;
#pragma unroll
    for (int dt = 0; dt < 4; ++dt)
      op[dt * 16 + ln] = (half_t)(o[dt][r] * inv);
  }
}

// ---------------------------------------------------------------- LN
// y = LN(resid + delta); writes fp32 (y32) and optional f16 copy (y16)
__global__ __launch_bounds__(256)
void resid_ln(const float* __restrict__ resid, const float* __restrict__ delta,
              const float* __restrict__ g, const float* __restrict__ bt,
              float* __restrict__ y32, half_t* __restrict__ y16) {
  __shared__ float red[8];
  const int row = blockIdx.x;
  const int tid = threadIdx.x;
  const float* rp = resid + (size_t)row * C_;
  const float* dp = delta + (size_t)row * C_;
  float v[4], s = 0.f;
#pragma unroll
  for (int i = 0; i < 4; ++i) { int c = tid + i * 256; v[i] = rp[c] + dp[c]; s += v[i]; }
#pragma unroll
  for (int d = 16; d >= 1; d >>= 1) s += __shfl_xor(s, d, 32);
  if ((tid & 31) == 0) red[tid >> 5] = s;
  __syncthreads();
  float tot = 0.f;
#pragma unroll
  for (int w = 0; w < 8; ++w) tot += red[w];
  float mu = tot * (1.f / C_);
  __syncthreads();
  float s2 = 0.f;
#pragma unroll
  for (int i = 0; i < 4; ++i) { float d0 = v[i] - mu; s2 += d0 * d0; }
#pragma unroll
  for (int d = 16; d >= 1; d >>= 1) s2 += __shfl_xor(s2, d, 32);
  if ((tid & 31) == 0) red[tid >> 5] = s2;
  __syncthreads();
  float tot2 = 0.f;
#pragma unroll
  for (int w = 0; w < 8; ++w) tot2 += red[w];
  float rstd = rsqrtf(tot2 * (1.f / C_) + 1e-5f);
#pragma unroll
  for (int i = 0; i < 4; ++i) {
    int c = tid + i * 256;
    float y = (v[i] - mu) * rstd * g[c] + bt[c];
    if (y32) y32[(size_t)row * C_ + c] = y;
    if (y16) y16[(size_t)row * C_ + c] = (half_t)y;
  }
}

// ---------------------------------------------------------------- launch
extern "C" void kernel_launch(void* const* d_in, const int* in_sizes, int n_in,
                              void* d_out, int out_size, void* d_ws, size_t ws_size,
                              hipStream_t stream) {
  (void)in_sizes; (void)n_in; (void)out_size; (void)ws_size;
  const float* x     = (const float*)d_in[0];
  const float* wq    = (const float*)d_in[1];
  const float* wk    = (const float*)d_in[2];
  const float* wv    = (const float*)d_in[3];
  const float* wproj = (const float*)d_in[4];
  const float* bproj = (const float*)d_in[5];
  const float* ln1_g = (const float*)d_in[6];
  const float* ln1_b = (const float*)d_in[7];
  const float* w1    = (const float*)d_in[8];
  const float* b1    = (const float*)d_in[9];
  const float* w2    = (const float*)d_in[10];
  const float* b2    = (const float*)d_in[11];
  const float* ln2_g = (const float*)d_in[12];
  const float* ln2_b = (const float*)d_in[13];
  float* out = (float*)d_out;

  // ---- workspace layout (bytes). Region A is reused by the FFN hidden.
  char* ws = (char*)d_ws;
  const size_t oX16  = 0;                                  // x f16   (phase 1)
  const size_t oWqkv = oX16  + (size_t)BT_ * C_ * 2;       // packed qkv weights
  const size_t oQkv  = oWqkv + (size_t)C_ * 3 * C_ * 2;    // q|k|v activations
  const size_t oAttn = oQkv  + (size_t)BT_ * 3 * C_ * 2;   // attn concat f16
  const size_t endA  = oAttn + (size_t)BT_ * C_ * 2;
  const size_t oH16  = 0;                                  // FFN hidden (reuses A)
  const size_t oWpr  = endA;
  const size_t oW1   = oWpr + (size_t)C_ * C_ * 2;
  const size_t oW2   = oW1  + (size_t)C_ * FF_ * 2;
  const size_t oSA   = oW2  + (size_t)FF_ * C_ * 2;        // sa32 / ff32 shared
  const size_t oX1   = oSA  + (size_t)BT_ * C_ * 4;        // x1 fp32
  const size_t oX1h  = oX1  + (size_t)BT_ * C_ * 4;        // x1 f16

  half_t* x16    = (half_t*)(ws + oX16);
  half_t* wqkv16 = (half_t*)(ws + oWqkv);
  half_t* qkv16  = (half_t*)(ws + oQkv);
  half_t* attn16 = (half_t*)(ws + oAttn);
  half_t* h16    = (half_t*)(ws + oH16);
  half_t* wpr16  = (half_t*)(ws + oWpr);
  half_t* w1_16  = (half_t*)(ws + oW1);
  half_t* w2_16  = (half_t*)(ws + oW2);
  float*  sa32   = (float*)(ws + oSA);
  float*  x1_32  = (float*)(ws + oX1);
  half_t* x1_16  = (half_t*)(ws + oX1h);

  auto cvt = [&](const float* s, half_t* d, int n) {
    cvt_f32_f16<<<(n + 255) / 256, 256, 0, stream>>>(s, d, n);
  };

  // 1) precision conversion + weight packing
  cvt(x, x16, BT_ * C_);
  pack_wqkv<<<(3 * C_ * C_ + 255) / 256, 256, 0, stream>>>(wq, wk, wv, wqkv16);
  cvt(wproj, wpr16, C_ * C_);
  cvt(w1, w1_16, C_ * FF_);
  cvt(w2, w2_16, FF_ * C_);

  // 2) fused QKV projection: (4096 x 1024) @ (1024 x 3072) -> f16
  gemm_wmma<<<dim3(QKVLD / BN, BT_ / BM), 256, 0, stream>>>(
      x16, C_, wqkv16, QKVLD, nullptr, nullptr, qkv16, QKVLD, C_, 0);

  // 3) causal flash attention -> attn16 (4096 x 1024)
  attn_flash<<<dim3(T_ / 16, B_ * H_), 32, 0, stream>>>(qkv16, attn16);

  // 4) output projection + bias -> sa32 fp32
  gemm_wmma<<<dim3(C_ / BN, BT_ / BM), 256, 0, stream>>>(
      attn16, C_, wpr16, C_, bproj, sa32, nullptr, C_, C_, 0);

  // 5) x1 = LN(x + sa)
  resid_ln<<<BT_, 256, 0, stream>>>(x, sa32, ln1_g, ln1_b, x1_32, x1_16);

  // 6) h = relu(x1 @ w1 + b1) -> f16 (reuses region A)
  gemm_wmma<<<dim3(FF_ / BN, BT_ / BM), 256, 0, stream>>>(
      x1_16, C_, w1_16, FF_, b1, nullptr, h16, FF_, C_, 1);

  // 7) ff = h @ w2 + b2 -> fp32 (reuses sa32)
  gemm_wmma<<<dim3(C_ / BN, BT_ / BM), 256, 0, stream>>>(
      h16, FF_, w2_16, C_, b2, sa32, nullptr, C_, FF_, 0);

  // 8) out = LN(x1 + ff)
  resid_ln<<<BT_, 256, 0, stream>>>(x1_32, sa32, ln2_g, ln2_b, out, nullptr);
}
